// EGNNLayer_30631706755112
// MI455X (gfx1250) — matrix operand order repaired and checked by reading, not compiled, not run on hardware
//
#include <hip/hip_runtime.h>

// EGNN-like equivariant layer for MI455X (gfx1250, wave32, WMMA).
// Edge half: wave-per-edge VALU + LDS weights + f32 atomics (L2-resident).
// Node half: per-node bilinear forms lowered to bf16 WMMA GEMMs
//            (outer products staged in LDS, weights pre-reshaped to bf16).

typedef __attribute__((ext_vector_type(16))) __bf16 v16bf;
typedef __attribute__((ext_vector_type(8)))  float  v8f;

__device__ __forceinline__ float sigmoidf_(float x) { return 1.f / (1.f + __expf(-x)); }
__device__ __forceinline__ float siluf_(float x)    { return x * sigmoidf_(x); }

union Frag16 { uint4 q[2]; v16bf v; };

__device__ __forceinline__ v16bf load_frag_pair(const __bf16* p0, const __bf16* p1) {
    Frag16 f;
    f.q[0] = *reinterpret_cast<const uint4*>(p0);
    f.q[1] = *reinterpret_cast<const uint4*>(p1);
    return f.v;
}

// ---------------------------------------------------------------------------
// Weight prep: reshape tp2 weights into bf16 B matrices, scales folded in.
// Bssvv: [w=0..95][k=0..5119]  k<4096: tp2_ss[u*64+v], else tp2_vv[u*32+v]
// Bsv  : [w=0..31][k=0..2047]  k = u*32+v  (tp2_sv)
// Bvs  : [w=0..31][k=0..2047]  k = u*64+v  (tp2_vs)
// Stored N(=w)-major with K contiguous -> B fragment = 2x 16B loads per lane.
// ---------------------------------------------------------------------------
__global__ void prep_weights(const float* __restrict__ tp2_ss,
                             const float* __restrict__ tp2_vv,
                             const float* __restrict__ tp2_sv,
                             const float* __restrict__ tp2_vs,
                             __bf16* __restrict__ Bssvv,
                             __bf16* __restrict__ Bsv,
                             __bf16* __restrict__ Bvs) {
    const float S_SS = 0.015625f;                  // 1/sqrt(64*64)
    const float S_VV = 0.57735026919f / 32.f;      // INV_SQRT3/sqrt(32*32)
    const float S_X  = 0.02209708691f;             // 1/sqrt(64*32)
    const int TOT0 = 96 * 5120;
    const int TOT1 = TOT0 + 32 * 2048;
    const int TOT2 = TOT1 + 32 * 2048;
    for (int i = blockIdx.x * blockDim.x + threadIdx.x; i < TOT2;
         i += gridDim.x * blockDim.x) {
        if (i < TOT0) {
            int w = i / 5120, k = i - w * 5120;
            float val = (k < 4096) ? tp2_ss[k * 96 + w] * S_SS
                                   : tp2_vv[(k - 4096) * 96 + w] * S_VV;
            Bssvv[i] = (__bf16)val;
        } else if (i < TOT1) {
            int j = i - TOT0; int w = j >> 11, k = j & 2047;
            Bsv[j] = (__bf16)(tp2_sv[k * 32 + w] * S_X);
        } else {
            int j = i - TOT1; int w = j >> 11, k = j & 2047;
            Bvs[j] = (__bf16)(tp2_vs[k * 32 + w] * S_X);
        }
    }
}

// ---------------------------------------------------------------------------
// Edge kernel: one wave (32 lanes) per edge. Shared tp1/lin1 weights in LDS.
// Lane l owns output channels {l, l+32, l+64} of ms and vector channel w=l.
// ---------------------------------------------------------------------------
__global__ void edge_kernel(const float* __restrict__ node_s,
                            const float* __restrict__ node_v,
                            const float* __restrict__ edge_sh,
                            const int*   __restrict__ edge_row,
                            const int*   __restrict__ edge_col,
                            const float* __restrict__ tp1_ss,
                            const float* __restrict__ tp1_vv,
                            const float* __restrict__ tp1_sv,
                            const float* __restrict__ tp1_vs,
                            const float* __restrict__ lin1_s,
                            const float* __restrict__ lin1_v,
                            float* __restrict__ agg_s,
                            float* __restrict__ agg_v,
                            int E, int iters) {
    __shared__ float Wss[64 * 96], Wvv[32 * 96], Wsv[64 * 32], Wvs[32 * 32];
    __shared__ float L1s[64 * 64], L1v[32 * 32];
    __shared__ float sj[8][64], vj[8][96], db[8][32], gb[8][64], mvb[8][96], esh[8][4];

    const int tid = threadIdx.x;
    for (int i = tid; i < 64 * 96; i += 256) Wss[i] = tp1_ss[i];
    for (int i = tid; i < 32 * 96; i += 256) Wvv[i] = tp1_vv[i];
    for (int i = tid; i < 64 * 32; i += 256) Wsv[i] = tp1_sv[i];
    for (int i = tid; i < 32 * 32; i += 256) Wvs[i] = tp1_vs[i];
    for (int i = tid; i < 64 * 64; i += 256) L1s[i] = lin1_s[i];
    for (int i = tid; i < 32 * 32; i += 256) L1v[i] = lin1_v[i];
    __syncthreads();

    const int wv = tid >> 5, l = tid & 31;
    const float INV3S = 0.57735026919f * 0.17677669529f; // INV_SQRT3/sqrt(32)
    const float IS32  = 0.17677669529f;                  // 1/sqrt(32)

    for (int it = 0; it < iters; ++it) {
        const int e = it * (int)gridDim.x * 8 + (int)blockIdx.x * 8 + wv;
        const bool valid = (e < E);
        int row = 0;
        if (valid) {
            row = edge_row[e];
            const int col = edge_col[e];
            const float* sp = node_s + (size_t)col * 64;
            const float* vp = node_v + (size_t)col * 96;
            sj[wv][l] = sp[l];         sj[wv][l + 32] = sp[l + 32];
            vj[wv][l] = vp[l];         vj[wv][l + 32] = vp[l + 32];
            vj[wv][l + 64] = vp[l + 64];
            if (l < 4) esh[wv][l] = edge_sh[(size_t)e * 4 + l];
            const int e2 = e + (int)gridDim.x * 8;
            if (e2 < E) __builtin_prefetch(edge_col + e2, 0, 1); // global_prefetch
        }
        __syncthreads();

        const float es  = esh[wv][0];
        const float ev0 = esh[wv][1], ev1 = esh[wv][2], ev2 = esh[wv][3];
        // per-channel dot v_j[u] . e_v   (u = lane)
        db[wv][l] = vj[wv][l * 3 + 0] * ev0 + vj[wv][l * 3 + 1] * ev1 +
                    vj[wv][l * 3 + 2] * ev2;
        __syncthreads();

        // ms[w] for w = l, l+32, l+64
        float a0 = 0.f, a1 = 0.f, a2 = 0.f;
        #pragma unroll 4
        for (int u = 0; u < 64; ++u) {
            const float su = sj[wv][u];
            a0 += su * Wss[u * 96 + l];
            a1 += su * Wss[u * 96 + l + 32];
            a2 += su * Wss[u * 96 + l + 64];
        }
        float b0 = 0.f, b1 = 0.f, b2 = 0.f;
        #pragma unroll 4
        for (int u = 0; u < 32; ++u) {
            const float du = db[wv][u];
            b0 += du * Wvv[u * 96 + l];
            b1 += du * Wvv[u * 96 + l + 32];
            b2 += du * Wvv[u * 96 + l + 64];
        }
        const float ms0 = es * a0 * 0.125f + b0 * INV3S;
        const float ms1 = es * a1 * 0.125f + b1 * INV3S;
        const float ms2 = es * a2 * 0.125f + b2 * INV3S;

        // tsv (w = l): (s_j @ Wsv)/8
        float tsv = 0.f;
        #pragma unroll 4
        for (int u = 0; u < 64; ++u) tsv += sj[wv][u] * Wsv[u * 32 + l];
        tsv *= 0.125f;

        // tvs (w = l, per c): e_s * (v_j[:,c] @ Wvs)/sqrt(32)
        float tv0 = 0.f, tv1 = 0.f, tv2 = 0.f;
        #pragma unroll 4
        for (int u = 0; u < 32; ++u) {
            const float w_ = Wvs[u * 32 + l];
            tv0 += vj[wv][u * 3 + 0] * w_;
            tv1 += vj[wv][u * 3 + 1] * w_;
            tv2 += vj[wv][u * 3 + 2] * w_;
        }
        const float fvs = es * IS32;

        // gate
        const float gt = sigmoidf_(ms2);
        gb[wv][l]      = siluf_(ms0);
        gb[wv][l + 32] = siluf_(ms1);
        mvb[wv][l * 3 + 0] = (tsv * ev0 + tv0 * fvs) * gt;
        mvb[wv][l * 3 + 1] = (tsv * ev1 + tv1 * fvs) * gt;
        mvb[wv][l * 3 + 2] = (tsv * ev2 + tv2 * fvs) * gt;
        __syncthreads();

        // lin1 + scatter
        float o0 = 0.f, o1 = 0.f;
        #pragma unroll 4
        for (int u = 0; u < 64; ++u) {
            const float g = gb[wv][u];
            o0 += g * L1s[u * 64 + l];
            o1 += g * L1s[u * 64 + l + 32];
        }
        float p0 = 0.f, p1 = 0.f, p2 = 0.f;
        #pragma unroll 4
        for (int u = 0; u < 32; ++u) {
            const float w_ = L1v[u * 32 + l];
            p0 += mvb[wv][u * 3 + 0] * w_;
            p1 += mvb[wv][u * 3 + 1] * w_;
            p2 += mvb[wv][u * 3 + 2] * w_;
        }
        if (valid) {
            float* as = agg_s + (size_t)row * 64;
            float* av = agg_v + (size_t)row * 96;
            atomicAdd(as + l,          o0 * 0.125f);
            atomicAdd(as + l + 32,     o1 * 0.125f);
            atomicAdd(av + l * 3 + 0,  p0 * IS32);
            atomicAdd(av + l * 3 + 1,  p1 * IS32);
            atomicAdd(av + l * 3 + 2,  p2 * IS32);
        }
        __syncthreads();
    }
}

// ---------------------------------------------------------------------------
// Node kernel: 16 nodes/block, 6 waves. Bilinear FCTP2 via bf16 WMMA:
//   y = A @ B  with A[n,k] = per-node outer products (built in LDS, K-chunked)
// Phase 1: ss+vv  K=5120, 96 cols (6 waves x one 16x16 C tile)
// Phase 2: sv     K=2048 per c (3 c x 2 col-tiles = 6 wave tasks)
// Phase 3: vs     K=2048 per c, accumulated into yv
// Epilogue: gate + lin2 + residual (VALU).
// ---------------------------------------------------------------------------
__global__ void node_kernel(const float* __restrict__ node_s,
                            const float* __restrict__ node_v,
                            const float* __restrict__ agg_s,
                            const float* __restrict__ agg_v,
                            const __bf16* __restrict__ Bssvv,
                            const __bf16* __restrict__ Bsv,
                            const __bf16* __restrict__ Bvs,
                            const float* __restrict__ lin2_s,
                            const float* __restrict__ lin2_v,
                            float* __restrict__ out_s,
                            float* __restrict__ out_v,
                            int N) {
    __shared__ float s1[16][64], s2[16][64], v1[16][96], v2[16][96];
    __shared__ __align__(16) __bf16 Abuf[3 * 16 * 1024];   // 96 KB, reused per phase
    __shared__ float ys[16][96];
    __shared__ float yv[16][96];
    __shared__ float L2s[64 * 64], L2v[32 * 32];

    const int tid = threadIdx.x;
    const int n0  = blockIdx.x * 16;

    for (int i = tid; i < 64 * 64; i += 192) L2s[i] = lin2_s[i];
    for (int i = tid; i < 32 * 32; i += 192) L2v[i] = lin2_v[i];
    for (int i = tid; i < 16 * 64; i += 192) {
        int n = i >> 6, w = i & 63, nd = n0 + n;
        s1[n][w] = (nd < N) ? node_s[(size_t)nd * 64 + w] : 0.f;
        s2[n][w] = (nd < N) ? agg_s[(size_t)nd * 64 + w] : 0.f;
    }
    for (int i = tid; i < 16 * 96; i += 192) {
        int n = i / 96, j = i - n * 96, nd = n0 + n;
        v1[n][j] = (nd < N) ? node_v[(size_t)nd * 96 + j] : 0.f;
        v2[n][j] = (nd < N) ? agg_v[(size_t)nd * 96 + j] : 0.f;
    }

    const int wv   = tid >> 5, l = tid & 31;
    const int lrow = l & 15;
    const int khiA = (l < 16) ? 0 : 16;   // A frag: K runs {0..15} / {16..31}
    const int khiB = (l < 16) ? 0 : 16;   // B frag: K runs {0..15} / {16..31}
    const int mb   = (l < 16) ? 0 : 8;    // C frag: M = r (+8 for upper half)

    // ---------------- Phase 1: ss + vv -> ys[16][96] ----------------
    v8f acc = {0.f, 0.f, 0.f, 0.f, 0.f, 0.f, 0.f, 0.f};
    const int K0s[3] = {0, 2048, 4096};
    const int KLs[3] = {2048, 2048, 1024};
    for (int ch = 0; ch < 3; ++ch) {
        const int K0 = K0s[ch], KL = KLs[ch];
        __syncthreads();   // prior-phase reads / staging done before overwrite
        for (int i = tid; i < 16 * KL; i += 192) {
            int n = i / KL, kk = i - n * KL, k = K0 + kk;
            float val;
            if (k < 4096) {
                int u = k >> 6, v = k & 63;
                val = s1[n][u] * s2[n][v];
            } else {
                int kv = k - 4096, u = kv >> 5, v = kv & 31;
                val = v1[n][u * 3 + 0] * v2[n][v * 3 + 0] +
                      v1[n][u * 3 + 1] * v2[n][v * 3 + 1] +
                      v1[n][u * 3 + 2] * v2[n][v * 3 + 2];
            }
            Abuf[n * 2048 + kk] = (__bf16)val;
        }
        __syncthreads();
        const __bf16* Bb = Bssvv + (size_t)(wv * 16 + lrow) * 5120 + K0 + khiB;
        const __bf16* Ab = Abuf + lrow * 2048 + khiA;
        for (int kk0 = 0; kk0 < KL; kk0 += 32) {
            v16bf a = load_frag_pair(Ab + kk0, Ab + kk0 + 8);
            v16bf b = load_frag_pair(Bb + kk0, Bb + kk0 + 8);
            acc = __builtin_amdgcn_wmma_f32_16x16x32_bf16(
                false, a, false, b, (short)0, acc, false, false);
        }
    }
    {
        const int w0 = wv * 16;
        #pragma unroll
        for (int r = 0; r < 8; ++r) ys[mb + r][w0 + lrow] = acc[r];
    }

    // ---------------- Phase 2: sv -> yv (write) ----------------
    const int c = wv >> 1, h = wv & 1;
    v8f acc2 = {0.f, 0.f, 0.f, 0.f, 0.f, 0.f, 0.f, 0.f};
    for (int ch = 0; ch < 2; ++ch) {
        const int K0 = ch * 1024;
        __syncthreads();
        for (int i = tid; i < 3 * 16 * 1024; i += 192) {
            int cc = i >> 14, n = (i >> 10) & 15, kk = i & 1023;
            int k = K0 + kk, u = k >> 5, v = k & 31;
            Abuf[i] = (__bf16)(s1[n][u] * v2[n][v * 3 + cc]);
        }
        __syncthreads();
        const __bf16* Bb = Bsv + (size_t)(h * 16 + lrow) * 2048 + K0 + khiB;
        const __bf16* Ab = Abuf + c * 16384 + lrow * 1024 + khiA;
        for (int kk0 = 0; kk0 < 1024; kk0 += 32) {
            v16bf a = load_frag_pair(Ab + kk0, Ab + kk0 + 8);
            v16bf b = load_frag_pair(Bb + kk0, Bb + kk0 + 8);
            acc2 = __builtin_amdgcn_wmma_f32_16x16x32_bf16(
                false, a, false, b, (short)0, acc2, false, false);
        }
    }
    #pragma unroll
    for (int r = 0; r < 8; ++r) yv[mb + r][(h * 16 + lrow) * 3 + c] = acc2[r];

    // ---------------- Phase 3: vs -> yv (accumulate) ----------------
    v8f acc3 = {0.f, 0.f, 0.f, 0.f, 0.f, 0.f, 0.f, 0.f};
    for (int ch = 0; ch < 2; ++ch) {
        const int K0 = ch * 1024;
        __syncthreads();
        for (int i = tid; i < 3 * 16 * 1024; i += 192) {
            int cc = i >> 14, n = (i >> 10) & 15, kk = i & 1023;
            int k = K0 + kk, u = k >> 6, v = k & 63;
            Abuf[i] = (__bf16)(v1[n][u * 3 + cc] * s2[n][v]);
        }
        __syncthreads();
        const __bf16* Bb = Bvs + (size_t)(h * 16 + lrow) * 2048 + K0 + khiB;
        const __bf16* Ab = Abuf + c * 16384 + lrow * 1024 + khiA;
        for (int kk0 = 0; kk0 < 1024; kk0 += 32) {
            v16bf a = load_frag_pair(Ab + kk0, Ab + kk0 + 8);
            v16bf b = load_frag_pair(Bb + kk0, Bb + kk0 + 8);
            acc3 = __builtin_amdgcn_wmma_f32_16x16x32_bf16(
                false, a, false, b, (short)0, acc3, false, false);
        }
    }
    __syncthreads();   // yv writes from phase 2 visible before accumulation
    #pragma unroll
    for (int r = 0; r < 8; ++r) yv[mb + r][(h * 16 + lrow) * 3 + c] += acc3[r];
    __syncthreads();

    // ---------------- Epilogue: gate + lin2 + residual ----------------
    for (int i = tid; i < 16 * 96; i += 192) {
        int n = i / 96, w = i - n * 96;
        float x = ys[n][w];
        ys[n][w] = (w < 64) ? siluf_(x) : sigmoidf_(x);
    }
    __syncthreads();
    for (int i = tid; i < 16 * 96; i += 192) {
        int n = i / 96, j = i - n * 96, w = j / 3;
        yv[n][j] *= ys[n][64 + w];
    }
    __syncthreads();
    for (int i = tid; i < 16 * 64; i += 192) {
        int n = i >> 6, w = i & 63, nd = n0 + n;
        if (nd < N) {
            float a = 0.f;
            #pragma unroll 4
            for (int u = 0; u < 64; ++u) a += ys[n][u] * L2s[u * 64 + w];
            out_s[(size_t)nd * 64 + w] = node_s[(size_t)nd * 64 + w] + a * 0.125f;
        }
    }
    for (int i = tid; i < 16 * 96; i += 192) {
        int n = i / 96, j = i - n * 96, w = j / 3, cc = j - w * 3, nd = n0 + n;
        if (nd < N) {
            float a = 0.f;
            #pragma unroll 4
            for (int u = 0; u < 32; ++u) a += yv[n][u * 3 + cc] * L2v[u * 32 + w];
            out_v[(size_t)nd * 96 + j] =
                node_v[(size_t)nd * 96 + j] + a * 0.17677669529f;
        }
    }
}

// ---------------------------------------------------------------------------
extern "C" void kernel_launch(void* const* d_in, const int* in_sizes, int n_in,
                              void* d_out, int out_size, void* d_ws, size_t ws_size,
                              hipStream_t stream) {
    const float* node_s  = (const float*)d_in[0];
    const float* node_v  = (const float*)d_in[1];
    const float* edge_sh = (const float*)d_in[2];
    const int*   edge_row = (const int*)d_in[4];
    const int*   edge_col = (const int*)d_in[5];
    const float* tp1_ss = (const float*)d_in[6];
    const float* tp1_vv = (const float*)d_in[7];
    const float* tp1_sv = (const float*)d_in[8];
    const float* tp1_vs = (const float*)d_in[9];
    const float* lin1_s = (const float*)d_in[10];
    const float* lin1_v = (const float*)d_in[11];
    const float* tp2_ss = (const float*)d_in[12];
    const float* tp2_vv = (const float*)d_in[13];
    const float* tp2_sv = (const float*)d_in[14];
    const float* tp2_vs = (const float*)d_in[15];
    const float* lin2_s = (const float*)d_in[16];
    const float* lin2_v = (const float*)d_in[17];

    const int N = in_sizes[0] / 64;
    const int E = in_sizes[4];

    char* ws = (char*)d_ws;
    size_t off = 0;
    auto take = [&](size_t bytes) -> char* {
        char* p = ws + off;
        off = (off + bytes + 255) & ~(size_t)255;
        return p;
    };
    float*  agg_s = (float*)take((size_t)N * 64 * sizeof(float));
    float*  agg_v = (float*)take((size_t)N * 96 * sizeof(float));
    __bf16* Bssvv = (__bf16*)take((size_t)96 * 5120 * 2);
    __bf16* Bsv   = (__bf16*)take((size_t)32 * 2048 * 2);
    __bf16* Bvs   = (__bf16*)take((size_t)32 * 2048 * 2);

    hipMemsetAsync(agg_s, 0, (size_t)N * 64 * sizeof(float), stream);
    hipMemsetAsync(agg_v, 0, (size_t)N * 96 * sizeof(float), stream);

    {
        int total  = 96 * 5120 + 32 * 2048 + 32 * 2048;
        int blocks = (total + 255) / 256;
        prep_weights<<<blocks, 256, 0, stream>>>(tp2_ss, tp2_vv, tp2_sv, tp2_vs,
                                                 Bssvv, Bsv, Bvs);
    }
    {
        int blocks = 4096;                                   // 8 waves = 8 edges/block
        int iters  = (E + blocks * 8 - 1) / (blocks * 8);
        edge_kernel<<<blocks, 256, 0, stream>>>(node_s, node_v, edge_sh,
                                                edge_row, edge_col,
                                                tp1_ss, tp1_vv, tp1_sv, tp1_vs,
                                                lin1_s, lin1_v,
                                                agg_s, agg_v, E, iters);
    }
    {
        int blocks = (N + 15) / 16;
        float* o_s = (float*)d_out;
        float* o_v = o_s + (size_t)N * 64;
        node_kernel<<<blocks, 192, 0, stream>>>(node_s, node_v, agg_s, agg_v,
                                                Bssvv, Bsv, Bvs,
                                                lin2_s, lin2_v, o_s, o_v, N);
    }
}
